// ModelRNN_67070209294845
// MI455X (gfx1250) — matrix operand order repaired and checked by reading, not compiled, not run on hardware
//
#include <hip/hip_runtime.h>
#include <hip/hip_bf16.h>

#define H 1024
#define S 256
#define VOUT 32000
#define KOUT 4096

typedef __attribute__((ext_vector_type(16))) __bf16        v16bf;
typedef __attribute__((ext_vector_type(8)))  float         v8f;
typedef __attribute__((ext_vector_type(4)))  float         v4f;
typedef __attribute__((ext_vector_type(4)))  unsigned int  v4u;

union ABFrag { v16bf bf; v4u u[2]; };

__device__ __forceinline__ void load_a_frag_f32(ABFrag& fa, const float* __restrict__ ap) {
    v4f f0 = *(const v4f*)(ap);
    v4f f1 = *(const v4f*)(ap + 4);
    v4f f2 = *(const v4f*)(ap + 16);
    v4f f3 = *(const v4f*)(ap + 20);
    fa.bf[0]  = (__bf16)f0[0]; fa.bf[1]  = (__bf16)f0[1]; fa.bf[2]  = (__bf16)f0[2]; fa.bf[3]  = (__bf16)f0[3];
    fa.bf[4]  = (__bf16)f1[0]; fa.bf[5]  = (__bf16)f1[1]; fa.bf[6]  = (__bf16)f1[2]; fa.bf[7]  = (__bf16)f1[3];
    fa.bf[8]  = (__bf16)f2[0]; fa.bf[9]  = (__bf16)f2[1]; fa.bf[10] = (__bf16)f2[2]; fa.bf[11] = (__bf16)f2[3];
    fa.bf[12] = (__bf16)f3[0]; fa.bf[13] = (__bf16)f3[1]; fa.bf[14] = (__bf16)f3[2]; fa.bf[15] = (__bf16)f3[3];
}

// C[m][n] = (ACCUM ? C[m][n] : 0) + (HAS_BIAS ? bias[m] : 0) + sum_k A_f32[m][k] * Bt_bf16[n][k]
// A row-major (lda), Bt row-major N x K (ldb), C row-major (ldc). N fixed = 256.
// grid.x = M/32, blockDim = 128 (4 waves, wave w covers columns w*64 .. w*64+63).
// Requires: M % 32 == 0, K % 32 == 0, ldb % 8 == 0, lda % 4 == 0.
template <bool HAS_BIAS, bool ACCUM>
__global__ void __launch_bounds__(128)
k_gemm(const float* __restrict__ A, int lda,
       const __bf16* __restrict__ Bt, int ldb,
       float* __restrict__ C, int ldc,
       const float* __restrict__ bias, int K)
{
    const int lane = threadIdx.x & 31;
    const int wave = threadIdx.x >> 5;
    const int ml   = lane & 15;
    const int half = lane >> 4;
    const int m0   = blockIdx.x * 32;
    const int n0   = wave * 64;

    v8f zero = {0.f,0.f,0.f,0.f,0.f,0.f,0.f,0.f};
    v8f acc[2][4];
#pragma unroll
    for (int mt = 0; mt < 2; ++mt)
#pragma unroll
        for (int nt = 0; nt < 4; ++nt) acc[mt][nt] = zero;

    for (int k0 = 0; k0 < K; k0 += 32) {
        ABFrag a[2];
#pragma unroll
        for (int mt = 0; mt < 2; ++mt) {
            const float* ap = A + (size_t)(m0 + mt*16 + ml) * (size_t)lda + (size_t)(k0 + half*8);
            load_a_frag_f32(a[mt], ap);
        }
        ABFrag b[4];
#pragma unroll
        for (int nt = 0; nt < 4; ++nt) {
            const __bf16* bp = Bt + (size_t)(n0 + nt*16 + ml) * (size_t)ldb + (size_t)(k0 + half*8);
            b[nt].u[0] = *(const v4u*)(bp);
            b[nt].u[1] = *(const v4u*)(bp + 16);
        }
#pragma unroll
        for (int mt = 0; mt < 2; ++mt)
#pragma unroll
            for (int nt = 0; nt < 4; ++nt)
                acc[mt][nt] = __builtin_amdgcn_wmma_f32_16x16x32_bf16(
                    false, a[mt].bf, false, b[nt].bf, (short)0, acc[mt][nt], false, false);
    }

#pragma unroll
    for (int mt = 0; mt < 2; ++mt) {
#pragma unroll
        for (int r = 0; r < 8; ++r) {
            const int m = m0 + mt*16 + half*8 + r;
            const float bv = HAS_BIAS ? bias[m] : 0.f;
            float* cp = C + (size_t)m * (size_t)ldc + (size_t)(n0 + ml);
#pragma unroll
            for (int nt = 0; nt < 4; ++nt) {
                float v = acc[mt][nt][r] + bv;
                if (ACCUM) v += cp[nt * 16];
                cp[nt * 16] = v;
            }
        }
    }
}

__global__ void k_zero(float* p, int n) {
    int i = blockIdx.x * blockDim.x + threadIdx.x;
    if (i < n) p[i] = 0.f;
}

// Gather embedding row and store as bf16.  grid = S blocks, 256 threads.
__global__ void k_gather(const float* __restrict__ emb, const int* __restrict__ src,
                         const int* __restrict__ tgt, __bf16* __restrict__ out, int dec)
{
    int t = blockIdx.x;
    int tok = dec ? (t == 0 ? src[0] : tgt[t-1]) : src[t];
    const float* e = emb + (size_t)tok * H;
    for (int k = threadIdx.x; k < H; k += blockDim.x)
        out[(size_t)t * H + k] = (__bf16)e[k];
}

// Generic wave-per-row matvec: y[r] = (b ? b[r] : 0) + sum_k W[r*cols+k] * x[k]
// grid.x * 8 waves == rows;  cols % 64 == 0.  blockDim = 256.
__global__ void k_matvec(const float* __restrict__ W, const float* __restrict__ x,
                         const float* __restrict__ b, float* __restrict__ y, int cols)
{
    int lane = threadIdx.x & 31;
    int row  = blockIdx.x * (blockDim.x >> 5) + (threadIdx.x >> 5);
    const float2* w2 = (const float2*)(W + (size_t)row * cols);
    const float2* x2 = (const float2*)x;
    int iters = cols >> 6;
    float a = 0.f;
    for (int it = 0; it < iters; ++it) {
        int idx = lane + it * 32;
        float2 w = w2[idx], v = x2[idx];
        a += w.x * v.x + w.y * v.y;
    }
    for (int m = 16; m >= 1; m >>= 1) a += __shfl_xor(a, m, 32);
    if (lane == 0) y[row] = a + (b ? b[row] : 0.f);
}

// One encoder GRU step, both directions.  grid = (128, 2), blockDim = 256 (wave per j).
__global__ void k_enc_step(const float* __restrict__ Whh_f, const float* __restrict__ Whh_b,
                           const float* __restrict__ bhh_f, const float* __restrict__ bhh_b,
                           const float* __restrict__ Gi_f,  const float* __restrict__ Gi_b,
                           const float* __restrict__ hf_in, float* __restrict__ hf_out,
                           const float* __restrict__ hb_in, float* __restrict__ hb_out,
                           float* __restrict__ enc_outs, __bf16* __restrict__ enc_outs_bf, int t)
{
    int lane = threadIdx.x & 31;
    int j    = blockIdx.x * (blockDim.x >> 5) + (threadIdx.x >> 5);
    int dir  = blockIdx.y;
    const float* W   = dir ? Whh_b : Whh_f;
    const float* bhh = dir ? bhh_b : bhh_f;
    const float* Gi  = dir ? Gi_b  : Gi_f;
    const float* hin = dir ? hb_in : hf_in;
    float*      hout = dir ? hb_out : hf_out;

    const float2* h2 = (const float2*)hin;
    const float2* Wr = (const float2*)(W + (size_t)(0*H + j) * H);
    const float2* Wz = (const float2*)(W + (size_t)(1*H + j) * H);
    const float2* Wn = (const float2*)(W + (size_t)(2*H + j) * H);

    float ar = 0.f, az = 0.f, an = 0.f;
    for (int it = 0; it < 16; ++it) {
        int idx = lane + it * 32;
        float2 h = h2[idx];
        float2 wr = Wr[idx]; ar += wr.x * h.x + wr.y * h.y;
        float2 wz = Wz[idx]; az += wz.x * h.x + wz.y * h.y;
        float2 wn = Wn[idx]; an += wn.x * h.x + wn.y * h.y;
    }
    for (int m = 16; m >= 1; m >>= 1) {
        ar += __shfl_xor(ar, m, 32);
        az += __shfl_xor(az, m, 32);
        an += __shfl_xor(an, m, 32);
    }
    if (lane == 0) {
        float ghr = ar + bhh[j], ghz = az + bhh[H + j], ghn = an + bhh[2*H + j];
        float gir = Gi[(size_t)(0*H + j) * S + t];
        float giz = Gi[(size_t)(1*H + j) * S + t];
        float gin = Gi[(size_t)(2*H + j) * S + t];
        float r = 1.f / (1.f + __expf(-(gir + ghr)));
        float z = 1.f / (1.f + __expf(-(giz + ghz)));
        float n = tanhf(gin + r * ghn);
        float hn = (1.f - z) * n + z * hin[j];
        hout[j] = hn;
        enc_outs[(size_t)t * (2*H) + dir*H + j]    = hn;
        enc_outs_bf[(size_t)t * (2*H) + dir*H + j] = (__bf16)hn;
    }
}

// Decoder attention: energy = tanh(h.v_dec + enc_score); attn = softmax(energy).
// One block of 256 threads.
__global__ void k_dec_attn(const float* __restrict__ h, const float* __restrict__ att_v,
                           const float* __restrict__ enc_score,
                           float* __restrict__ attn_f, __bf16* __restrict__ attn_bf, int t)
{
    __shared__ float sh[256];
    int tid = threadIdx.x;
    float p = 0.f;
    for (int k = tid; k < H; k += 256) p += h[k] * att_v[k];
    sh[tid] = p; __syncthreads();
    for (int s = 128; s > 0; s >>= 1) { if (tid < s) sh[tid] += sh[tid + s]; __syncthreads(); }
    float sdot = sh[0]; __syncthreads();

    float e = tanhf(sdot + enc_score[tid]);
    sh[tid] = e; __syncthreads();
    for (int s = 128; s > 0; s >>= 1) { if (tid < s) sh[tid] = fmaxf(sh[tid], sh[tid + s]); __syncthreads(); }
    float mx = sh[0]; __syncthreads();

    float ex = __expf(e - mx);
    sh[tid] = ex; __syncthreads();
    for (int s = 128; s > 0; s >>= 1) { if (tid < s) sh[tid] += sh[tid + s]; __syncthreads(); }
    float a = ex / sh[0];

    attn_f[(size_t)t * S + tid]  = a;
    attn_bf[(size_t)t * S + tid] = (__bf16)a;
}

// Decoder GRU step.  gi = Gi_e[:,t] (+bih folded) + W2E @ attn_t; gh = Whh @ h (+bhh).
// grid = 128, blockDim = 256 (wave per j).
__global__ void k_dec_step(const float* __restrict__ Whh, const float* __restrict__ bhh,
                           const float* __restrict__ Gi_e, const float* __restrict__ W2E,
                           const float* __restrict__ attn_t,
                           const float* __restrict__ h_in, float* __restrict__ h_out,
                           __bf16* __restrict__ Hbf, int t)
{
    int lane = threadIdx.x & 31;
    int j    = blockIdx.x * (blockDim.x >> 5) + (threadIdx.x >> 5);

    float sr = 0.f, sz = 0.f, sng = 0.f, snh = 0.f;

    // W2E part: rows of length 256 (f32) dotted with attn_t
    const float2* a2  = (const float2*)attn_t;
    const float2* Er  = (const float2*)(W2E + (size_t)(0*H + j) * S);
    const float2* Ez  = (const float2*)(W2E + (size_t)(1*H + j) * S);
    const float2* En  = (const float2*)(W2E + (size_t)(2*H + j) * S);
    for (int it = 0; it < 4; ++it) {
        int idx = lane + it * 32;
        float2 a = a2[idx];
        float2 w = Er[idx]; sr  += w.x * a.x + w.y * a.y;
        float2 u = Ez[idx]; sz  += u.x * a.x + u.y * a.y;
        float2 v = En[idx]; sng += v.x * a.x + v.y * a.y;
    }
    // Whh part: rows of length 1024 (f32) dotted with h
    const float2* h2 = (const float2*)h_in;
    const float2* Wr = (const float2*)(Whh + (size_t)(0*H + j) * H);
    const float2* Wz = (const float2*)(Whh + (size_t)(1*H + j) * H);
    const float2* Wn = (const float2*)(Whh + (size_t)(2*H + j) * H);
    for (int it = 0; it < 16; ++it) {
        int idx = lane + it * 32;
        float2 h = h2[idx];
        float2 w = Wr[idx]; sr  += w.x * h.x + w.y * h.y;
        float2 u = Wz[idx]; sz  += u.x * h.x + u.y * h.y;
        float2 v = Wn[idx]; snh += v.x * h.x + v.y * h.y;
    }
    for (int m = 16; m >= 1; m >>= 1) {
        sr  += __shfl_xor(sr,  m, 32);
        sz  += __shfl_xor(sz,  m, 32);
        sng += __shfl_xor(sng, m, 32);
        snh += __shfl_xor(snh, m, 32);
    }
    if (lane == 0) {
        float sR = sr + Gi_e[(size_t)(0*H + j) * S + t] + bhh[j];
        float sZ = sz + Gi_e[(size_t)(1*H + j) * S + t] + bhh[H + j];
        float gN = sng + Gi_e[(size_t)(2*H + j) * S + t];
        float hN = snh + bhh[2*H + j];
        float r = 1.f / (1.f + __expf(-sR));
        float z = 1.f / (1.f + __expf(-sZ));
        float n = tanhf(gN + r * hN);
        float hn = (1.f - z) * n + z * h_in[j];
        h_out[j] = hn;
        Hbf[(size_t)t * H + j] = (__bf16)hn;
    }
}

// Per-timestep log_softmax over V from logits stored m-major (VOUT x S).
// grid = S, blockDim = 256.
__global__ void k_logsoftmax(const float* __restrict__ logits, float* __restrict__ out)
{
    __shared__ float sh[256];
    int t = blockIdx.x, tid = threadIdx.x;
    float mx = -1e30f;
    for (int m = tid; m < VOUT; m += 256) mx = fmaxf(mx, logits[(size_t)m * S + t]);
    sh[tid] = mx; __syncthreads();
    for (int s = 128; s > 0; s >>= 1) { if (tid < s) sh[tid] = fmaxf(sh[tid], sh[tid + s]); __syncthreads(); }
    float M = sh[0]; __syncthreads();
    float sum = 0.f;
    for (int m = tid; m < VOUT; m += 256) sum += __expf(logits[(size_t)m * S + t] - M);
    sh[tid] = sum; __syncthreads();
    for (int s = 128; s > 0; s >>= 1) { if (tid < s) sh[tid] += sh[tid + s]; __syncthreads(); }
    float lse = M + __logf(sh[0]);
    for (int m = tid; m < VOUT; m += 256)
        out[(size_t)t * VOUT + m] = logits[(size_t)m * S + t] - lse;
}

extern "C" void kernel_launch(void* const* d_in, const int* in_sizes, int n_in,
                              void* d_out, int out_size, void* d_ws, size_t ws_size,
                              hipStream_t stream) {
    const int*   src       = (const int*)  d_in[0];
    const int*   tgt       = (const int*)  d_in[1];
    const float* enc_emb   = (const float*)d_in[2];
    const float* enc_Wih_f = (const float*)d_in[3];
    const float* enc_Whh_f = (const float*)d_in[4];
    const float* enc_bih_f = (const float*)d_in[5];
    const float* enc_bhh_f = (const float*)d_in[6];
    const float* enc_Wih_b = (const float*)d_in[7];
    const float* enc_Whh_b = (const float*)d_in[8];
    const float* enc_bih_b = (const float*)d_in[9];
    const float* enc_bhh_b = (const float*)d_in[10];
    const float* enc_fc_W  = (const float*)d_in[11];
    const float* enc_fc_b  = (const float*)d_in[12];
    const float* dec_emb   = (const float*)d_in[13];
    const float* dec_Wih   = (const float*)d_in[14];
    const float* dec_Whh   = (const float*)d_in[15];
    const float* dec_bih   = (const float*)d_in[16];
    const float* dec_bhh   = (const float*)d_in[17];
    const float* att_v     = (const float*)d_in[18];
    const float* out_W     = (const float*)d_in[19];
    const float* out_b     = (const float*)d_in[20];
    float* out = (float*)d_out;

    // ---- workspace carve-out (256B aligned) ----
    char* base = (char*)d_ws;
    size_t off = 0;
    auto alloc = [&](size_t bytes) -> void* {
        void* p = base + off;
        off = (off + bytes + 255) & ~(size_t)255;
        return p;
    };
    __bf16* x_emb_bf   = (__bf16*)alloc((size_t)S * H * 2);        // encoder input emb (bf16)
    __bf16* E_bf       = (__bf16*)alloc((size_t)S * H * 2);        // decoder input emb (bf16)
    float*  Gi_f       = (float*) alloc((size_t)3*H * S * 4);
    float*  Gi_b       = (float*) alloc((size_t)3*H * S * 4);
    float*  Gi_e       = (float*) alloc((size_t)3*H * S * 4);
    float*  enc_outs   = (float*) alloc((size_t)S * 2*H * 4);
    __bf16* enc_out_bf = (__bf16*)alloc((size_t)S * 2*H * 2);
    float*  hf_buf     = (float*) alloc((size_t)2 * H * 4);        // ping-pong
    float*  hb_buf     = (float*) alloc((size_t)2 * H * 4);
    float*  hd_buf     = (float*) alloc((size_t)2 * H * 4);
    float*  enc_score  = (float*) alloc((size_t)S * 4);
    float*  attn_f     = (float*) alloc((size_t)S * S * 4);
    __bf16* attn_bf    = (__bf16*)alloc((size_t)S * S * 2);
    __bf16* H_bf       = (__bf16*)alloc((size_t)S * H * 2);
    float*  W2E        = (float*) alloc((size_t)3*H * S * 4);      // dec_Wih[:,H:3H] @ enc_outs^T
    float*  OWCE       = (float*) alloc((size_t)VOUT * S * 4);     // out_W[:,H:3H] @ enc_outs^T
    float*  logits     = (float*) alloc((size_t)VOUT * S * 4);     // m-major VOUT x S
    (void)ws_size; (void)n_in; (void)in_sizes; (void)out_size;

    // 0) zero encoder hidden-state ping-pong buffers
    k_zero<<<16, 256, 0, stream>>>(hf_buf, 2 * H);
    k_zero<<<16, 256, 0, stream>>>(hb_buf, 2 * H);

    // 1) embedding gathers (f32 -> bf16)
    k_gather<<<S, 256, 0, stream>>>(enc_emb, src, tgt, x_emb_bf, 0);
    k_gather<<<S, 256, 0, stream>>>(dec_emb, src, tgt, E_bf, 1);

    // 2) batched input-gate GEMMs: Gi = Wih @ X^T + bih   (M=3072, K=1024, N=256)
    k_gemm<true,  false><<<3*H/32, 128, 0, stream>>>(enc_Wih_f, H, x_emb_bf, H, Gi_f, S, enc_bih_f, H);
    k_gemm<true,  false><<<3*H/32, 128, 0, stream>>>(enc_Wih_b, H, x_emb_bf, H, Gi_b, S, enc_bih_b, H);
    k_gemm<true,  false><<<3*H/32, 128, 0, stream>>>(dec_Wih, 3*H, E_bf, H, Gi_e, S, dec_bih, H);

    // 3) sequential encoder (both directions per step)
    for (int t = 0; t < S; ++t) {
        const float* fi = hf_buf + (t & 1) * H;  float* fo = hf_buf + ((t + 1) & 1) * H;
        const float* bi = hb_buf + (t & 1) * H;  float* bo = hb_buf + ((t + 1) & 1) * H;
        k_enc_step<<<dim3(128, 2), 256, 0, stream>>>(enc_Whh_f, enc_Whh_b, enc_bhh_f, enc_bhh_b,
                                                     Gi_f, Gi_b, fi, fo, bi, bo,
                                                     enc_outs, enc_out_bf, t);
    }
    const float* hbT = hb_buf;  // after 256 steps, final state is in slot 0

    // 4) enc_score = enc_outs @ v_enc ; dec_h0 = enc_fc_W @ hb_T + enc_fc_b
    k_matvec<<<S/8, 256, 0, stream>>>(enc_outs, att_v + H, (const float*)nullptr, enc_score, 2*H);
    k_matvec<<<H/8, 256, 0, stream>>>(enc_fc_W, hbT, enc_fc_b, hd_buf, H);

    // 5) attention hoisting GEMMs over enc_outs (K=2048, N=256)
    k_gemm<false, false><<<3*H/32, 128, 0, stream>>>(dec_Wih + H, 3*H, enc_out_bf, 2*H, W2E, S,
                                                     (const float*)nullptr, 2*H);
    k_gemm<false, false><<<VOUT/32, 128, 0, stream>>>(out_W + H, KOUT, enc_out_bf, 2*H, OWCE, S,
                                                      (const float*)nullptr, 2*H);

    // 6) sequential decoder
    for (int t = 0; t < S; ++t) {
        const float* hi = hd_buf + (t & 1) * H;
        float*       ho = hd_buf + ((t + 1) & 1) * H;
        k_dec_attn<<<1, 256, 0, stream>>>(hi, att_v, enc_score, attn_f, attn_bf, t);
        k_dec_step<<<128, 256, 0, stream>>>(dec_Whh, dec_bhh, Gi_e, W2E,
                                            attn_f + (size_t)t * S, hi, ho, H_bf, t);
    }

    // 7) logits = out_W[:, :H] @ E^T + out_b  +  OWCE @ Attn^T  +  out_W[:, 3H:] @ H^T
    k_gemm<true,  false><<<VOUT/32, 128, 0, stream>>>(out_W, KOUT, E_bf, H, logits, S, out_b, H);
    k_gemm<false, true ><<<VOUT/32, 128, 0, stream>>>(OWCE, S, attn_bf, S, logits, S,
                                                      (const float*)nullptr, S);
    k_gemm<false, true ><<<VOUT/32, 128, 0, stream>>>(out_W + 3*H, KOUT, H_bf, H, logits, S,
                                                      (const float*)nullptr, H);

    // 8) log_softmax -> d_out (S x VOUT)
    k_logsoftmax<<<S, 256, 0, stream>>>(logits, out);
}
